// LSTMEncoder_4492535792368
// MI455X (gfx1250) — matrix operand order repaired and checked by reading, not compiled, not run on hardware
//
#include <hip/hip_runtime.h>
#include <hip/hip_bf16.h>

// ---------------------------------------------------------------------------
// 2-layer LSTM, N=128, L=512, IN=128, H=1024. bf16 WMMA (16x16x32, f32 acc).
// Each wave: 16 rows x 16 hidden units x all 4 gates, fused gate nonlinearity
// and c/h update. Weights converted to bf16 once; they fit in L2 (192MB).
// ---------------------------------------------------------------------------

#define NB    128
#define LSEQ  512
#define INDIM 128
#define HDIM  1024
#define G4H   (4 * HDIM)

typedef __attribute__((ext_vector_type(16))) __bf16 bf16x16;
typedef __attribute__((ext_vector_type(8)))  float  f32x8;

union FragCast { uint4 u[2]; bf16x16 v; };

// A fragment: 16x32 bf16 tile, `tile` points at (row0, k0), stride in elems.
// lane m = lane&15 ; half = lane>>4 selects K sub-chunks per ISA layout:
//   half=0: K 0..7 and 16..23 ; half=1: K 8..15 and 24..31
__device__ __forceinline__ bf16x16 ld_fragA(const __hip_bfloat16* tile,
                                            int stride, int lane) {
    int m = lane & 15, h = lane >> 4;
    const __hip_bfloat16* p = tile + (size_t)m * stride + h * 8;
    FragCast c;
    c.u[0] = *(const uint4*)(p);
    c.u[1] = *(const uint4*)(p + 16);
    return c.v;
}

// B fragment: 32x16 (KxN). B(k,n) = W[n][k] with W row-major, so each lane
// reads two runs of 8 contiguous bf16 (p, p+16) of its W row. `tile` points at (wrow0, k0).
//   lane n = lane&15 ; half=0 -> K 0..7 and 16..23, half=1 -> K 8..15 and 24..31
__device__ __forceinline__ bf16x16 ld_fragB(const __hip_bfloat16* tile,
                                            int stride, int lane) {
    int n = lane & 15, h = lane >> 4;
    const __hip_bfloat16* p = tile + (size_t)n * stride + h * 8;
    FragCast c;
    c.u[0] = *(const uint4*)(p);
    c.u[1] = *(const uint4*)(p + 16);
    return c.v;
}

__device__ __forceinline__ float sigm(float x) {
    return 1.0f / (1.0f + __expf(-x));
}

// Fused LSTM step: gates = xA @ Wih^T + hA @ Whh^T + bias; c,h update.
// Grid: 128 blocks x 128 threads (4 waves). Wave -> 16 rows x 16 units x 4 gates.
__global__ void __launch_bounds__(128)
lstm_step_kernel(const __hip_bfloat16* __restrict__ xA, int x_stride, int Kx,
                 const __hip_bfloat16* __restrict__ Wih,   // [4H x Kx] row-major
                 const __hip_bfloat16* __restrict__ hA,    // [128 x H] bf16
                 const __hip_bfloat16* __restrict__ Whh,   // [4H x H] row-major
                 const float* __restrict__ bias,           // [4H] (b_ih + b_hh)
                 float* __restrict__ cst,                  // [128 x H] cell state
                 __hip_bfloat16* __restrict__ h_out,       // [128 x H] bf16
                 float* __restrict__ h_out_f32)            // optional f32 out
{
    const int lane  = threadIdx.x & 31;
    const int wave  = threadIdx.x >> 5;
    const int mbase = (blockIdx.x & 7) << 4;                    // 0..112
    const int j0    = ((blockIdx.x >> 3) << 6) + (wave << 4);   // 0..1008

    f32x8 acc0 = {}, acc1 = {}, acc2 = {}, acc3 = {};  // i, f, g, o

    // ---- x_t contribution -------------------------------------------------
    for (int k = 0; k < Kx; k += 32) {
        bf16x16 a = ld_fragA(xA + (size_t)mbase * x_stride + k, x_stride, lane);
        bf16x16 b0 = ld_fragB(Wih + (size_t)(0 * HDIM + j0) * Kx + k, Kx, lane);
        acc0 = __builtin_amdgcn_wmma_f32_16x16x32_bf16(false, a, false, b0,
                                                       (short)0, acc0, false, false);
        bf16x16 b1 = ld_fragB(Wih + (size_t)(1 * HDIM + j0) * Kx + k, Kx, lane);
        acc1 = __builtin_amdgcn_wmma_f32_16x16x32_bf16(false, a, false, b1,
                                                       (short)0, acc1, false, false);
        bf16x16 b2 = ld_fragB(Wih + (size_t)(2 * HDIM + j0) * Kx + k, Kx, lane);
        acc2 = __builtin_amdgcn_wmma_f32_16x16x32_bf16(false, a, false, b2,
                                                       (short)0, acc2, false, false);
        bf16x16 b3 = ld_fragB(Wih + (size_t)(3 * HDIM + j0) * Kx + k, Kx, lane);
        acc3 = __builtin_amdgcn_wmma_f32_16x16x32_bf16(false, a, false, b3,
                                                       (short)0, acc3, false, false);
    }

    // ---- h_{t-1} contribution --------------------------------------------
    for (int k = 0; k < HDIM; k += 32) {
        bf16x16 a = ld_fragA(hA + (size_t)mbase * HDIM + k, HDIM, lane);
        bf16x16 b0 = ld_fragB(Whh + (size_t)(0 * HDIM + j0) * HDIM + k, HDIM, lane);
        acc0 = __builtin_amdgcn_wmma_f32_16x16x32_bf16(false, a, false, b0,
                                                       (short)0, acc0, false, false);
        bf16x16 b1 = ld_fragB(Whh + (size_t)(1 * HDIM + j0) * HDIM + k, HDIM, lane);
        acc1 = __builtin_amdgcn_wmma_f32_16x16x32_bf16(false, a, false, b1,
                                                       (short)0, acc1, false, false);
        bf16x16 b2 = ld_fragB(Whh + (size_t)(2 * HDIM + j0) * HDIM + k, HDIM, lane);
        acc2 = __builtin_amdgcn_wmma_f32_16x16x32_bf16(false, a, false, b2,
                                                       (short)0, acc2, false, false);
        bf16x16 b3 = ld_fragB(Whh + (size_t)(3 * HDIM + j0) * HDIM + k, HDIM, lane);
        acc3 = __builtin_amdgcn_wmma_f32_16x16x32_bf16(false, a, false, b3,
                                                       (short)0, acc3, false, false);
    }

    // ---- gate nonlinearities + state update (in-register) -----------------
    const int j     = j0 + (lane & 15);
    const int rbase = mbase + ((lane >> 4) << 3);
    const float bi = bias[0 * HDIM + j];
    const float bf = bias[1 * HDIM + j];
    const float bg = bias[2 * HDIM + j];
    const float bo = bias[3 * HDIM + j];

#pragma unroll
    for (int v = 0; v < 8; ++v) {
        const int r  = rbase + v;
        const size_t idx = (size_t)r * HDIM + j;
        float iv = sigm(acc0[v] + bi);
        float fv = sigm(acc1[v] + bf);
        float gv = tanhf(acc2[v] + bg);
        float ov = sigm(acc3[v] + bo);
        float cn = fv * cst[idx] + iv * gv;
        cst[idx] = cn;
        float hv = ov * tanhf(cn);
        h_out[idx] = __float2bfloat16(hv);
        if (h_out_f32) h_out_f32[idx] = hv;
    }
}

// ---------------------------------------------------------------------------
// Helper kernels
// ---------------------------------------------------------------------------
__global__ void cvt_f32_to_bf16(const float* __restrict__ in,
                                __hip_bfloat16* __restrict__ out, int n) {
    int i = blockIdx.x * blockDim.x + threadIdx.x;
    int s = gridDim.x * blockDim.x;
    for (; i < n; i += s) out[i] = __float2bfloat16(in[i]);
}

__global__ void bias_combine(const float* __restrict__ a,
                             const float* __restrict__ b,
                             float* __restrict__ out, int n) {
    int i = blockIdx.x * blockDim.x + threadIdx.x;
    if (i < n) out[i] = a[i] + b[i];
}

__global__ void zero_u32(unsigned* __restrict__ p, int n) {
    int i = blockIdx.x * blockDim.x + threadIdx.x;
    int s = gridDim.x * blockDim.x;
    for (; i < n; i += s) p[i] = 0u;
}

// ---------------------------------------------------------------------------
extern "C" void kernel_launch(void* const* d_in, const int* in_sizes, int n_in,
                              void* d_out, int out_size, void* d_ws, size_t ws_size,
                              hipStream_t stream) {
    const float* x    = (const float*)d_in[0];
    const float* Wih0 = (const float*)d_in[1];
    const float* Whh0 = (const float*)d_in[2];
    const float* bih0 = (const float*)d_in[3];
    const float* bhh0 = (const float*)d_in[4];
    const float* Wih1 = (const float*)d_in[5];
    const float* Whh1 = (const float*)d_in[6];
    const float* bih1 = (const float*)d_in[7];
    const float* bhh1 = (const float*)d_in[8];

    char* ws = (char*)d_ws;
    size_t off = 0;
    auto carve = [&](size_t bytes) -> void* {
        void* p = ws + off;
        off = (off + bytes + 255) & ~(size_t)255;
        return p;
    };

    const int nX    = NB * LSEQ * INDIM;       // 8.4M elems
    const int nWih0 = G4H * INDIM;
    const int nWhh  = G4H * HDIM;
    const int nHC   = NB * HDIM;               // 131072 elems

    __hip_bfloat16* xb     = (__hip_bfloat16*)carve((size_t)nX * 2);
    __hip_bfloat16* Wih0b  = (__hip_bfloat16*)carve((size_t)nWih0 * 2);
    __hip_bfloat16* Whh0b  = (__hip_bfloat16*)carve((size_t)nWhh * 2);
    __hip_bfloat16* Wih1b  = (__hip_bfloat16*)carve((size_t)nWhh * 2);
    __hip_bfloat16* Whh1b  = (__hip_bfloat16*)carve((size_t)nWhh * 2);
    float*          bias0  = (float*)carve((size_t)G4H * 4);
    float*          bias1  = (float*)carve((size_t)G4H * 4);
    __hip_bfloat16* h0buf0 = (__hip_bfloat16*)carve((size_t)nHC * 2);
    __hip_bfloat16* h0buf1 = (__hip_bfloat16*)carve((size_t)nHC * 2);
    __hip_bfloat16* h1buf0 = (__hip_bfloat16*)carve((size_t)nHC * 2);
    __hip_bfloat16* h1buf1 = (__hip_bfloat16*)carve((size_t)nHC * 2);
    float*          c0     = (float*)carve((size_t)nHC * 4);
    float*          c1     = (float*)carve((size_t)nHC * 4);

    __hip_bfloat16* h0buf[2] = {h0buf0, h0buf1};
    __hip_bfloat16* h1buf[2] = {h1buf0, h1buf1};

    // --- one-time conversions (graph-capturable, deterministic) -----------
    cvt_f32_to_bf16<<<1024, 256, 0, stream>>>(x, xb, nX);
    cvt_f32_to_bf16<<<512, 256, 0, stream>>>(Wih0, Wih0b, nWih0);
    cvt_f32_to_bf16<<<1024, 256, 0, stream>>>(Whh0, Whh0b, nWhh);
    cvt_f32_to_bf16<<<1024, 256, 0, stream>>>(Wih1, Wih1b, nWhh);
    cvt_f32_to_bf16<<<1024, 256, 0, stream>>>(Whh1, Whh1b, nWhh);
    bias_combine<<<(G4H + 255) / 256, 256, 0, stream>>>(bih0, bhh0, bias0, G4H);
    bias_combine<<<(G4H + 255) / 256, 256, 0, stream>>>(bih1, bhh1, bias1, G4H);

    // --- state init (must happen every call; ws is poisoned once) ---------
    zero_u32<<<256, 256, 0, stream>>>((unsigned*)c0, nHC);
    zero_u32<<<256, 256, 0, stream>>>((unsigned*)c1, nHC);
    zero_u32<<<256, 256, 0, stream>>>((unsigned*)h0buf0, nHC / 2);
    zero_u32<<<256, 256, 0, stream>>>((unsigned*)h0buf1, nHC / 2);
    zero_u32<<<256, 256, 0, stream>>>((unsigned*)h1buf0, nHC / 2);
    zero_u32<<<256, 256, 0, stream>>>((unsigned*)h1buf1, nHC / 2);

    // --- sequential recurrence: 2 fused step kernels per time step --------
    dim3 sgrid(128), sblock(128);
    for (int l = 0; l < LSEQ; ++l) {
        __hip_bfloat16* h0_in  = h0buf[l & 1];
        __hip_bfloat16* h0_out = h0buf[(l + 1) & 1];
        lstm_step_kernel<<<sgrid, sblock, 0, stream>>>(
            xb + (size_t)l * INDIM, LSEQ * INDIM, INDIM,
            Wih0b, h0_in, Whh0b, bias0, c0, h0_out, nullptr);

        __hip_bfloat16* h1_in  = h1buf[l & 1];
        __hip_bfloat16* h1_out = h1buf[(l + 1) & 1];
        lstm_step_kernel<<<sgrid, sblock, 0, stream>>>(
            h0_out, HDIM, HDIM,
            Wih1b, h1_in, Whh1b, bias1, c1, h1_out,
            (l == LSEQ - 1) ? (float*)d_out : nullptr);
    }
}
